// ONIDecoder_30837865185770
// MI455X (gfx1250) — matrix operand order repaired and checked by reading, not compiled, not run on hardware
//
#include <hip/hip_runtime.h>
#include <hip/hip_bf16.h>
#include <math.h>

// ---------------------------------------------------------------------------
// ONIDecoder (LSTM head) for MI455X / gfx1250, wave32, bf16 WMMA 16x16x32,
// triple-buffered LDS with distance-2 async global->LDS prefetch (ASYNCcnt).
// ---------------------------------------------------------------------------

typedef __bf16 bf16;
typedef __attribute__((ext_vector_type(8)))  bf16  v8bf;
typedef __attribute__((ext_vector_type(16))) bf16  v16bf;
typedef __attribute__((ext_vector_type(8)))  float v8f;

#define B_      512
#define ENC_    2048
#define DEC_    1024
#define TSTEPS_ 12
#define PRED_   24
#define IN_DIM_ 2061
#define IN_PAD  2080           // 2061 padded to multiple of 32
#define HS_K    (DEC_ * PRED_) // 24576
#define FOURD   (4 * DEC_)     // 4096

// ---- CDNA5 async global->LDS copy (GLOBAL_LOAD_ASYNC_TO_LDS_B128) ---------
#if defined(__has_builtin)
#if __has_builtin(__builtin_amdgcn_global_load_async_to_lds_b128)
#define HAVE_ASYNC_LDS 1
#endif
#endif
#ifndef HAVE_ASYNC_LDS
#define HAVE_ASYNC_LDS 0
#endif

typedef int v4i __attribute__((ext_vector_type(4)));
typedef __attribute__((address_space(1))) v4i gv4i_t;   // global int4
typedef __attribute__((address_space(3))) v4i lv4i_t;   // LDS int4

static __device__ __forceinline__ void async_copy_b128(void* lds_dst, const void* gsrc) {
#if HAVE_ASYNC_LDS
  __builtin_amdgcn_global_load_async_to_lds_b128(
      (gv4i_t*)(uintptr_t)gsrc,
      (lv4i_t*)(uint32_t)(uintptr_t)lds_dst, 0, 0);
#else
  (void)lds_dst; (void)gsrc;
#endif
}

// Wait until ASYNCcnt <= N. Async loads complete in order, so N = (#copies of
// the younger in-flight group) guarantees the older group has fully landed.
template <int N>
static __device__ __forceinline__ void async_wait_le() {
#if HAVE_ASYNC_LDS
#if __has_builtin(__builtin_amdgcn_s_wait_asynccnt)
  __builtin_amdgcn_s_wait_asynccnt(N);
#else
  asm volatile("s_wait_asynccnt %0" :: "i"(N) : "memory");
#endif
#endif
}

static __device__ __forceinline__ float sigf(float x) {
  return 1.0f / (1.0f + __expf(-x));
}

static __device__ __forceinline__ v16bf cat2(v8bf lo, v8bf hi) {
  return __builtin_shufflevector(lo, hi, 0,1,2,3,4,5,6,7,8,9,10,11,12,13,14,15);
}

static __device__ __forceinline__ v8f vzero8() {
  v8f z = {0.f,0.f,0.f,0.f,0.f,0.f,0.f,0.f};
  return z;
}

// --------------------------- small prep kernels ----------------------------

__global__ void pack_x_kernel(const float* __restrict__ enc,
                              const float* __restrict__ oni,
                              const float* __restrict__ ssi,
                              bf16* __restrict__ xb) {
  int id = blockIdx.x * blockDim.x + threadIdx.x;
  if (id >= B_ * IN_PAD) return;
  int b = id / IN_PAD, k = id - b * IN_PAD;
  float v = 0.f;
  if (k < ENC_)            v = enc[(size_t)b * ENC_ + k];
  else if (k == ENC_)      v = oni[b];
  else if (k < IN_DIM_)    v = ssi[(size_t)b * TSTEPS_ + (k - ENC_ - 1)];
  xb[(size_t)b * IN_PAD + k] = (bf16)v;
}

__global__ void convert_pad_kernel(const float* __restrict__ src,
                                   bf16* __restrict__ dst,
                                   int N, int K, int Kpad) {
  int groups = Kpad >> 3;
  int id = blockIdx.x * blockDim.x + threadIdx.x;
  if (id >= N * groups) return;
  int n = id / groups;
  int c = (id - n * groups) << 3;
  v8bf out;
#pragma unroll
  for (int j = 0; j < 8; ++j) {
    int k = c + j;
    out[j] = (k < K) ? (bf16)src[(size_t)n * K + k] : (bf16)0.f;
  }
  *(v8bf*)(dst + (size_t)n * Kpad + c) = out;
}

__global__ void init_state_kernel(bf16* __restrict__ h, float* __restrict__ c) {
  int id = blockIdx.x * blockDim.x + threadIdx.x;
  if (id >= B_ * DEC_) return;
  h[id] = (bf16)0.f;
  c[id] = 0.f;
}

// ----------------------- generic WMMA GEMM: X @ W^T ------------------------
// C(M x N) = A(M x Kpad) @ Bm(N x Kpad)^T + bias0 (+ bias1); act==1: sigmoid.
// Block tile 64x64, 8 waves each 16(M) x 32(N). Triple-buffered LDS, one
// barrier per 32-wide K slab; slab k+2 streams in while slab k feeds WMMA.
__global__ __launch_bounds__(256)
void gemm_bt_kernel(const bf16* __restrict__ A, int lda,
                    const bf16* __restrict__ Bm, int ldb,
                    const float* __restrict__ bias0,
                    const float* __restrict__ bias1,
                    float* __restrict__ outF, bf16* __restrict__ outB,
                    int act, int N, int Kpad) {
  __shared__ bf16 As[3][64][40];   // 80B row stride: conflict-free, 16B-aligned
  __shared__ bf16 Bs[3][64][40];
  const int tid  = threadIdx.x;
  const int lane = tid & 31, wave = tid >> 5;
  const int wm = wave & 3, wn = wave >> 2;
  const int mblk = blockIdx.y * 64, nblk = blockIdx.x * 64;
  const int lr = tid >> 2, lc = (tid & 3) << 3;   // fill: 16B per thread/tile
  const bf16* aP = A  + (size_t)(mblk + lr) * lda + lc;
  const bf16* bP = Bm + (size_t)(nblk + lr) * ldb + lc;
  const int m16 = lane & 15, hi = lane >> 4;
  v8f acc0 = vzero8(), acc1 = vzero8();

  const int nk = Kpad >> 5;   // always >= 2 here
  int cur = 0;
#if HAVE_ASYNC_LDS
  async_copy_b128(&As[0][lr][lc], aP);
  async_copy_b128(&Bs[0][lr][lc], bP);
  async_copy_b128(&As[1][lr][lc], aP + 32);
  async_copy_b128(&Bs[1][lr][lc], bP + 32);
#else
  *(v8bf*)&As[0][lr][lc] = *(const v8bf*)aP;
  *(v8bf*)&Bs[0][lr][lc] = *(const v8bf*)bP;
#endif

  for (int kt = 0; kt < nk; ++kt) {
#if HAVE_ASYNC_LDS
    if (kt + 1 < nk) async_wait_le<2>();   // group kt done, kt+1 in flight
    else             async_wait_le<0>();   // last slab: full drain
    __syncthreads();
    if (kt + 2 < nk) {                     // stream slab kt+2
      int nxt2 = cur + 2; if (nxt2 >= 3) nxt2 -= 3;
      async_copy_b128(&As[nxt2][lr][lc], aP + (kt + 2) * 32);
      async_copy_b128(&Bs[nxt2][lr][lc], bP + (kt + 2) * 32);
    }
#else
    v8bf pva, pvb;
    const bool pf = (kt + 1 < nk);
    if (pf) {                              // register prefetch of slab kt+1
      pva = *(const v8bf*)(aP + (kt + 1) * 32);
      pvb = *(const v8bf*)(bP + (kt + 1) * 32);
    }
    __syncthreads();
#endif
    // A frag 16x32: lane half selects K 0-15 vs 16-31 (contiguous)
    const bf16* pa = &As[cur][wm * 16 + m16][hi * 16];
    v16bf af = cat2(*(const v8bf*)pa, *(const v8bf*)(pa + 8));
    // B frag 32x16: lane = N, lane half selects K 0-15 vs 16-31 (contiguous)
    const bf16* pb0 = &Bs[cur][wn * 32 + m16][hi * 16];
    v16bf bf0 = cat2(*(const v8bf*)pb0, *(const v8bf*)(pb0 + 8));
    const bf16* pb1 = &Bs[cur][wn * 32 + 16 + m16][hi * 16];
    v16bf bf1 = cat2(*(const v8bf*)pb1, *(const v8bf*)(pb1 + 8));
    acc0 = __builtin_amdgcn_wmma_f32_16x16x32_bf16(false, af, false, bf0,
                                                   (short)0, acc0, false, false);
    acc1 = __builtin_amdgcn_wmma_f32_16x16x32_bf16(false, af, false, bf1,
                                                   (short)0, acc1, false, false);
#if !HAVE_ASYNC_LDS
    if (pf) {
      int nxt1 = cur + 1; if (nxt1 >= 3) nxt1 -= 3;
      *(v8bf*)&As[nxt1][lr][lc] = pva;
      *(v8bf*)&Bs[nxt1][lr][lc] = pvb;
    }
#endif
    cur += 1; if (cur >= 3) cur -= 3;
  }

#pragma unroll
  for (int s = 0; s < 2; ++s) {
    v8f accv = s ? acc1 : acc0;
    int ng = nblk + wn * 32 + s * 16 + m16;
    float bsum = bias0[ng] + (bias1 ? bias1[ng] : 0.f);
#pragma unroll
    for (int v = 0; v < 8; ++v) {
      int mg = mblk + wm * 16 + hi * 8 + v;   // C/D layout: M = 8*hi + vgpr
      float r = accv[v] + bsum;
      if (act) r = sigf(r);
      size_t o = (size_t)mg * N + ng;
      if (outF) outF[o] = r;
      if (outB) outB[o] = (bf16)r;
    }
  }
}

// ------------------------- fused LSTM step kernel --------------------------
// gates = xproj + h_in @ W_hh^T (4 gate tiles per block), then elementwise
// LSTM update; writes h_out (bf16) and hs[b][d*24 + t] (bf16, (B, D*T) order).
__global__ __launch_bounds__(256)
void lstm_step_kernel(const bf16* __restrict__ hin,
                      const bf16* __restrict__ Whh,    // (4096 x 1024) bf16
                      const float* __restrict__ xproj, // (512 x 4096) incl. biases
                      float* __restrict__ cst,
                      bf16* __restrict__ hout,
                      bf16* __restrict__ hs, int t) {
  __shared__ bf16 As[3][64][40];
  __shared__ bf16 Bs[3][4][64][40];
  const int tid  = threadIdx.x;
  const int lane = tid & 31, wave = tid >> 5;
  const int wm = wave & 3, wn = wave >> 2;
  const int mblk = blockIdx.y * 64, nblk = blockIdx.x * 64;
  const int lr = tid >> 2, lc = (tid & 3) << 3;
  const int m16 = lane & 15, hi = lane >> 4;
  const bf16* aP = hin + (size_t)(mblk + lr) * DEC_ + lc;
  const bf16* bP[4];
#pragma unroll
  for (int g = 0; g < 4; ++g)
    bP[g] = Whh + (size_t)(g * DEC_ + nblk + lr) * DEC_ + lc;

  v8f acc[4][2];
#pragma unroll
  for (int g = 0; g < 4; ++g) { acc[g][0] = vzero8(); acc[g][1] = vzero8(); }

  const int nk = DEC_ >> 5;   // 32 slabs
  int cur = 0;
#if HAVE_ASYNC_LDS
  async_copy_b128(&As[0][lr][lc], aP);
#pragma unroll
  for (int g = 0; g < 4; ++g) async_copy_b128(&Bs[0][g][lr][lc], bP[g]);
  async_copy_b128(&As[1][lr][lc], aP + 32);
#pragma unroll
  for (int g = 0; g < 4; ++g) async_copy_b128(&Bs[1][g][lr][lc], bP[g] + 32);
#else
  *(v8bf*)&As[0][lr][lc] = *(const v8bf*)aP;
#pragma unroll
  for (int g = 0; g < 4; ++g) *(v8bf*)&Bs[0][g][lr][lc] = *(const v8bf*)bP[g];
#endif

  for (int kt = 0; kt < nk; ++kt) {
#if HAVE_ASYNC_LDS
    if (kt + 1 < nk) async_wait_le<5>();   // group kt done, kt+1 (5 copies) in flight
    else             async_wait_le<0>();
    __syncthreads();
    if (kt + 2 < nk) {
      int nxt2 = cur + 2; if (nxt2 >= 3) nxt2 -= 3;
      const int koff2 = (kt + 2) * 32;
      async_copy_b128(&As[nxt2][lr][lc], aP + koff2);
#pragma unroll
      for (int g = 0; g < 4; ++g)
        async_copy_b128(&Bs[nxt2][g][lr][lc], bP[g] + koff2);
    }
#else
    v8bf pva, pvb[4];
    const bool pf = (kt + 1 < nk);
    const int koff = (kt + 1) * 32;
    if (pf) {
      pva = *(const v8bf*)(aP + koff);
#pragma unroll
      for (int g = 0; g < 4; ++g) pvb[g] = *(const v8bf*)(bP[g] + koff);
    }
    __syncthreads();
#endif
    const bf16* pa = &As[cur][wm * 16 + m16][hi * 16];
    v16bf af = cat2(*(const v8bf*)pa, *(const v8bf*)(pa + 8));
#pragma unroll
    for (int g = 0; g < 4; ++g) {
      const bf16* pb0 = &Bs[cur][g][wn * 32 + m16][hi * 16];
      v16bf bf0 = cat2(*(const v8bf*)pb0, *(const v8bf*)(pb0 + 8));
      const bf16* pb1 = &Bs[cur][g][wn * 32 + 16 + m16][hi * 16];
      v16bf bf1 = cat2(*(const v8bf*)pb1, *(const v8bf*)(pb1 + 8));
      acc[g][0] = __builtin_amdgcn_wmma_f32_16x16x32_bf16(false, af, false, bf0,
                                                (short)0, acc[g][0], false, false);
      acc[g][1] = __builtin_amdgcn_wmma_f32_16x16x32_bf16(false, af, false, bf1,
                                                (short)0, acc[g][1], false, false);
    }
#if !HAVE_ASYNC_LDS
    if (pf) {
      int nxt1 = cur + 1; if (nxt1 >= 3) nxt1 -= 3;
      *(v8bf*)&As[nxt1][lr][lc] = pva;
#pragma unroll
      for (int g = 0; g < 4; ++g) *(v8bf*)&Bs[nxt1][g][lr][lc] = pvb[g];
    }
#endif
    cur += 1; if (cur >= 3) cur -= 3;
  }

#pragma unroll
  for (int s = 0; s < 2; ++s) {
    int ng = nblk + wn * 32 + s * 16 + m16;
#pragma unroll
    for (int v = 0; v < 8; ++v) {
      int mg = mblk + wm * 16 + hi * 8 + v;
      size_t xo = (size_t)mg * FOURD + ng;
      float gi = acc[0][s][v] + xproj[xo];
      float gf = acc[1][s][v] + xproj[xo + 1 * DEC_];
      float gg = acc[2][s][v] + xproj[xo + 2 * DEC_];
      float go = acc[3][s][v] + xproj[xo + 3 * DEC_];
      size_t co = (size_t)mg * DEC_ + ng;
      float cn = sigf(gf) * cst[co] + sigf(gi) * tanhf(gg);
      cst[co] = cn;
      float h = sigf(go) * tanhf(cn);
      hout[co] = (bf16)h;
      hs[(size_t)mg * HS_K + (size_t)ng * PRED_ + t] = (bf16)h;
    }
  }
}

// --------------------------------- launch ----------------------------------

extern "C" void kernel_launch(void* const* d_in, const int* in_sizes, int n_in,
                              void* d_out, int out_size, void* d_ws, size_t ws_size,
                              hipStream_t stream) {
  (void)in_sizes; (void)n_in; (void)out_size; (void)ws_size;
  const float* enc  = (const float*)d_in[0];
  const float* oni  = (const float*)d_in[1];
  const float* ssi  = (const float*)d_in[2];
  const float* W_ih = (const float*)d_in[3];
  const float* b_ih = (const float*)d_in[4];
  const float* W_hh = (const float*)d_in[5];
  const float* b_hh = (const float*)d_in[6];
  const float* W_l1 = (const float*)d_in[7];
  const float* b_l1 = (const float*)d_in[8];
  const float* W_l2 = (const float*)d_in[9];
  const float* b_l2 = (const float*)d_in[10];
  float* out = (float*)d_out;

  uint8_t* ws = (uint8_t*)d_ws;
  size_t off = 0;
  auto carve = [&](size_t bytes) -> void* {
    void* p = ws + off;
    off += (bytes + 255) & ~(size_t)255;
    return p;
  };
  bf16*  xb    = (bf16*) carve((size_t)B_ * IN_PAD * 2);
  bf16*  WihB  = (bf16*) carve((size_t)FOURD * IN_PAD * 2);
  bf16*  WhhB  = (bf16*) carve((size_t)FOURD * DEC_ * 2);
  bf16*  Wl1B  = (bf16*) carve((size_t)DEC_ * HS_K * 2);
  bf16*  Wl2B  = (bf16*) carve((size_t)DEC_ * DEC_ * 2);
  float* xproj = (float*)carve((size_t)B_ * FOURD * 4);
  bf16*  h0    = (bf16*) carve((size_t)B_ * DEC_ * 2);
  bf16*  h1    = (bf16*) carve((size_t)B_ * DEC_ * 2);
  float* cst   = (float*)carve((size_t)B_ * DEC_ * 4);
  bf16*  hsb   = (bf16*) carve((size_t)B_ * HS_K * 2);
  bf16*  p1    = (bf16*) carve((size_t)B_ * DEC_ * 2);

  pack_x_kernel<<<(B_ * IN_PAD + 255) / 256, 256, 0, stream>>>(enc, oni, ssi, xb);
  convert_pad_kernel<<<(FOURD * (IN_PAD / 8) + 255) / 256, 256, 0, stream>>>(
      W_ih, WihB, FOURD, IN_DIM_, IN_PAD);
  convert_pad_kernel<<<(FOURD * (DEC_ / 8) + 255) / 256, 256, 0, stream>>>(
      W_hh, WhhB, FOURD, DEC_, DEC_);
  convert_pad_kernel<<<(DEC_ * (HS_K / 8) + 255) / 256, 256, 0, stream>>>(
      W_l1, Wl1B, DEC_, HS_K, HS_K);
  convert_pad_kernel<<<(DEC_ * (DEC_ / 8) + 255) / 256, 256, 0, stream>>>(
      W_l2, Wl2B, DEC_, DEC_, DEC_);
  init_state_kernel<<<(B_ * DEC_ + 255) / 256, 256, 0, stream>>>(h0, cst);

  // x_proj = x @ W_ih^T + b_ih + b_hh   (512 x 4096, f32)
  gemm_bt_kernel<<<dim3(FOURD / 64, B_ / 64), 256, 0, stream>>>(
      xb, IN_PAD, WihB, IN_PAD, b_ih, b_hh, xproj, nullptr, 0, FOURD, IN_PAD);

  // 24 sequential LSTM steps (ping-pong h to avoid cross-block races)
  bf16* hp = h0; bf16* hq = h1;
  for (int t = 0; t < PRED_; ++t) {
    lstm_step_kernel<<<dim3(DEC_ / 64, B_ / 64), 256, 0, stream>>>(
        hp, WhhB, xproj, cst, hq, hsb, t);
    bf16* tmp = hp; hp = hq; hq = tmp;
  }

  // preds = sigmoid(hs @ W_l1^T + b_l1)   (512 x 1024, bf16)
  gemm_bt_kernel<<<dim3(DEC_ / 64, B_ / 64), 256, 0, stream>>>(
      hsb, HS_K, Wl1B, HS_K, b_l1, nullptr, nullptr, p1, 1, DEC_, HS_K);

  // out = sigmoid(preds @ W_l2^T + b_l2)  (512 x 1024, f32 -> d_out)
  gemm_bt_kernel<<<dim3(DEC_ / 64, B_ / 64), 256, 0, stream>>>(
      p1, DEC_, Wl2B, DEC_, b_l2, nullptr, out, nullptr, 1, DEC_, DEC_);
}
